// Attention_48352741818760
// MI455X (gfx1250) — compile-verified, matrix operand order checked by reading, not run on hardware
//
#include <hip/hip_runtime.h>
#include <hip/hip_fp16.h>
#include <math.h>
#include <stdint.h>

// ---------------------------------------------------------------------------
// Types for CDNA5 WMMA (gfx1250, wave32)
// ---------------------------------------------------------------------------
typedef __attribute__((ext_vector_type(16))) _Float16 v16h;
typedef __attribute__((ext_vector_type(8)))  _Float16 v8h;
typedef __attribute__((ext_vector_type(8)))  float    v8f;
typedef __attribute__((ext_vector_type(4)))  int      v4i;

#define N_NODES   1024
#define N_EDGES   3072
#define T_TOT     4096      // N_NODES + N_EDGES
#define D_MODEL   128
#define D_INNER   512
#define N_HEADS   8
#define D_HEAD    64
#define E_FEAT    64
#define E_DIM     320       // 2*D_MODEL + E_FEAT (combined edge GEMM K)
#define VT_LD     4096      // leading dim of transposed V [H][DH][T]

// constants: c = (1-0.8)*sqrt(2), s = 1/sqrt(2)  ->  s*c = 0.2 exactly
#define SCALE_S      0.70710678118654752f
#define SCALE_SC     0.20000000000000000f
#define SCALE_SALPHA 0.56568542494923802f   // s * 0.8
#define ATTN_SCALE   0.125f                 // DH^-0.5

// ---------------------------------------------------------------------------
// Async global->LDS copy (ISA: GLOBAL_LOAD_ASYNC_TO_LDS_B128, ASYNCcnt).
// Builtin signature (from hipcc diagnostic): (v4i AS1*, v4i AS3*, imm, imm).
// Generic->AS casts go through uintptr_t: on amdgcn the low 32 bits of a
// generic LDS address are the AS(3) offset; global generic == AS(1).
// ---------------------------------------------------------------------------
#if __has_builtin(__builtin_amdgcn_global_load_async_to_lds_b128)
#define ASYNC_CP16(g, l)                                                       \
  __builtin_amdgcn_global_load_async_to_lds_b128(                              \
      (__attribute__((address_space(1))) v4i*)(uintptr_t)(g),                  \
      (__attribute__((address_space(3))) v4i*)(uintptr_t)(l), 0, 0)
#if __has_builtin(__builtin_amdgcn_s_wait_asynccnt)
#define WAIT_ASYNC() __builtin_amdgcn_s_wait_asynccnt(0)
#else
#define WAIT_ASYNC() asm volatile("s_wait_asynccnt 0x0" ::: "memory")
#endif
#else
#define ASYNC_CP16(g, l) (*(v8h*)(l) = *(const v8h*)(g))
#define WAIT_ASYNC() ((void)0)
#endif

// ---------------------------------------------------------------------------
// 16x32 f16 fragment load (A-layout per ISA 7.12.2; used for both A and B:
// halves 0..7 -> K = k0 + 8*laneHi + j ; halves 8..15 -> K = k0+16+8*laneHi+j)
// base must point at the start of the row (A) / column (col-major B).
// ---------------------------------------------------------------------------
__device__ __forceinline__ v16h frag16(const _Float16* base, int k0, int laneHi) {
  const _Float16* p = base + k0 + laneHi * 8;
  v8h x0 = *(const v8h*)(p);
  v8h x1 = *(const v8h*)(p + 16);
  v16h r;
#pragma unroll
  for (int i = 0; i < 8; ++i) { r[i] = x0[i]; r[i + 8] = x1[i]; }
  return r;
}

__device__ __forceinline__ v8f wmma32(v16h a, v16h b, v8f c) {
  return __builtin_amdgcn_wmma_f32_16x16x32_f16(false, a, false, b, (short)0, c,
                                                false, false);
}

// ---------------------------------------------------------------------------
// Prep kernels
// ---------------------------------------------------------------------------
__global__ void k_cvt_f16(const float* __restrict__ s, _Float16* __restrict__ d, int n) {
  int i = blockIdx.x * blockDim.x + threadIdx.x;
  if (i < n) d[i] = (_Float16)s[i];
}

// dst[n*dstStride + dstOff + k] = scale * src[k*Ns + n]   (transpose + convert)
__global__ void k_transpose_cvt(const float* __restrict__ src, int K, int Ns,
                                _Float16* __restrict__ dst, int dstStride, int dstOff,
                                float scale) {
  int i = blockIdx.x * blockDim.x + threadIdx.x;
  if (i >= K * Ns) return;
  int k = i / Ns, n = i % Ns;
  dst[(size_t)n * dstStride + dstOff + k] = (_Float16)(scale * src[i]);
}

__global__ void k_bias2(const float* __restrict__ a, const float* __restrict__ b,
                        float sa, float sb, float* __restrict__ d, int n) {
  int i = blockIdx.x * blockDim.x + threadIdx.x;
  if (i < n) d[i] = sa * a[i] + sb * b[i];
}

// edges_h[e][0:128] = nodes[e0], [128:256] = nodes[e1], [256:320] = edge_features
__global__ void k_prep_edges(const float* __restrict__ nodes, const float* __restrict__ ef,
                             const int* __restrict__ eidx, _Float16* __restrict__ edges) {
  int i = blockIdx.x * blockDim.x + threadIdx.x;
  if (i >= N_EDGES * E_DIM) return;
  int e = i / E_DIM, c = i % E_DIM;
  float v;
  if (c < D_MODEL)            v = nodes[eidx[e] * D_MODEL + c];
  else if (c < 2 * D_MODEL)   v = nodes[eidx[N_EDGES + e] * D_MODEL + (c - D_MODEL)];
  else                        v = ef[e * E_FEAT + (c - 2 * D_MODEL)];
  edges[i] = (_Float16)v;
}

// ---------------------------------------------------------------------------
// Node projection GEMM: [1024 x 128] @ [128 x 512] + bias
// writes fp32 (for edge alpha-sums) and f16 head layouts (row and/or V^T).
// 8 waves / block, one 16x16 tile per wave.
// ---------------------------------------------------------------------------
__global__ void k_gemm_node(const _Float16* __restrict__ A, const _Float16* __restrict__ Wt,
                            const float* __restrict__ bias, float* __restrict__ outF,
                            _Float16* __restrict__ outHeadRow, _Float16* __restrict__ outHeadT) {
  int lane = threadIdx.x & 31, wave = threadIdx.x >> 5;
  int laneHi = lane >> 4, laneLo = lane & 15;
  int tile = blockIdx.x * 8 + wave;          // 64 * 32 = 2048 tiles
  int m0 = (tile >> 5) * 16, n0 = (tile & 31) * 16;
  const _Float16* arow = A  + (size_t)(m0 + laneLo) * D_MODEL;
  const _Float16* bcol = Wt + (size_t)(n0 + laneLo) * D_MODEL;
  v8f acc = {};
#pragma unroll
  for (int k0 = 0; k0 < D_MODEL; k0 += 32)
    acc = wmma32(frag16(arow, k0, laneHi), frag16(bcol, k0, laneHi), acc);
  int col = n0 + laneLo;
  float bi = bias[col];
  int h = col >> 6, dh = col & 63;
#pragma unroll
  for (int r = 0; r < 8; ++r) {
    int row = m0 + r + 8 * laneHi;
    float v = acc[r] + bi;
    outF[(size_t)row * D_INNER + col] = v;
    if (outHeadRow) outHeadRow[((size_t)h * T_TOT + row) * D_HEAD + dh] = (_Float16)v;
    if (outHeadT)   outHeadT[((size_t)h * D_HEAD + dh) * VT_LD + row]   = (_Float16)v;
  }
}

// ---------------------------------------------------------------------------
// Edge projection GEMM: [3072 x 320] @ [320 x 512] (+ combined bias
// + s*alpha*(x[e0]+x[e1]) gather epilogue). Tokens land at t = N_NODES + e.
// ---------------------------------------------------------------------------
__global__ void k_gemm_edge(const _Float16* __restrict__ A, const _Float16* __restrict__ Wc,
                            const float* __restrict__ biasC, const float* __restrict__ aSrc,
                            const int* __restrict__ eidx,
                            _Float16* __restrict__ outHeadRow, _Float16* __restrict__ outHeadT) {
  int lane = threadIdx.x & 31, wave = threadIdx.x >> 5;
  int laneHi = lane >> 4, laneLo = lane & 15;
  int tile = blockIdx.x * 8 + wave;          // 192 * 32 = 6144 tiles
  int m0 = (tile >> 5) * 16, n0 = (tile & 31) * 16;
  const _Float16* arow = A  + (size_t)(m0 + laneLo) * E_DIM;
  const _Float16* bcol = Wc + (size_t)(n0 + laneLo) * E_DIM;
  v8f acc = {};
#pragma unroll
  for (int k0 = 0; k0 < E_DIM; k0 += 32)
    acc = wmma32(frag16(arow, k0, laneHi), frag16(bcol, k0, laneHi), acc);
  int col = n0 + laneLo;
  float bi = biasC[col];
  int h = col >> 6, dh = col & 63;
#pragma unroll
  for (int r = 0; r < 8; ++r) {
    int e = m0 + r + 8 * laneHi;
    int i0 = eidx[e], i1 = eidx[N_EDGES + e];
    float v = acc[r] + bi +
              SCALE_SALPHA * (aSrc[(size_t)i0 * D_INNER + col] +
                              aSrc[(size_t)i1 * D_INNER + col]);
    int t = N_NODES + e;
    if (outHeadRow) outHeadRow[((size_t)h * T_TOT + t) * D_HEAD + dh] = (_Float16)v;
    if (outHeadT)   outHeadT[((size_t)h * D_HEAD + dh) * VT_LD + t]   = (_Float16)v;
  }
}

// ---------------------------------------------------------------------------
// Flash attention with async double-buffered K/V tile staging in LDS.
// Block = 8 waves; all waves share the same 32-key tile, so K (32x64 f16) and
// V^T (64x32 f16) tiles are DMA'd cooperatively: 256 threads x one
// global_load_async_to_lds_b128 each per tile, tracked by ASYNCcnt.
// Per wave: 16-row query strip, online softmax, P reshaped via LDS.
// ---------------------------------------------------------------------------
__global__ void k_attn(const _Float16* __restrict__ Qh, const _Float16* __restrict__ Kh,
                       const _Float16* __restrict__ Vt, _Float16* __restrict__ Oh) {
  __shared__ __align__(16) _Float16 Ktile[2][32][64];  // 8 KB, [key][dh]
  __shared__ __align__(16) _Float16 Vtile[2][64][32];  // 8 KB, [dh][key]
  __shared__ __align__(16) _Float16 Plds[8][16][32];   // 8 KB, per-wave P

  int tid = threadIdx.x;
  int lane = tid & 31, wave = tid >> 5;
  int laneHi = lane >> 4, laneLo = lane & 15;
  int h = blockIdx.y;
  int qm0 = blockIdx.x * 128 + wave * 16;
  const _Float16* Q  = Qh + (size_t)h * T_TOT * D_HEAD;
  const _Float16* Kp = Kh + (size_t)h * T_TOT * D_HEAD;
  const _Float16* Vp = Vt + (size_t)h * D_HEAD * VT_LD;

  // cooperative 16B-chunk assignments (256 chunks per tile, 256 threads)
  int krow = tid >> 3, kch = (tid & 7) * 8;   // K tile: 32 rows x 64 halves
  int vrow = tid >> 2, vch = (tid & 3) * 8;   // V tile: 64 rows x 32 halves

  const _Float16* qrow = Q + (size_t)(qm0 + laneLo) * D_HEAD;
  v16h aq0 = frag16(qrow, 0, laneHi);
  v16h aq1 = frag16(qrow, 32, laneHi);

  v8f o0 = {}, o1 = {}, o2 = {}, o3 = {};
  float mr[8], lr[8];
#pragma unroll
  for (int r = 0; r < 8; ++r) { mr[r] = -1e30f; lr[r] = 0.0f; }

  // preload first K/V tile into buffer 0
  ASYNC_CP16(Kp + (size_t)krow * D_HEAD + kch, &Ktile[0][krow][kch]);
  ASYNC_CP16(Vp + (size_t)vrow * VT_LD + vch,  &Vtile[0][vrow][vch]);

  for (int j0 = 0; j0 < T_TOT; j0 += 32) {
    int buf = (j0 >> 5) & 1;
    WAIT_ASYNC();        // own async loads done (ASYNCcnt == 0)
    __syncthreads();     // join: every wave's chunks have landed in LDS
    if (j0 + 32 < T_TOT) {  // issue next tile into the other buffer
      ASYNC_CP16(Kp + (size_t)(j0 + 32 + krow) * D_HEAD + kch, &Ktile[buf ^ 1][krow][kch]);
      ASYNC_CP16(Vp + (size_t)vrow * VT_LD + (j0 + 32) + vch,  &Vtile[buf ^ 1][vrow][vch]);
    }

    // S tile (16 rows x 32 keys): 4 WMMAs, K fragments from LDS
    const _Float16* kc0 = &Ktile[buf][laneLo][0];
    const _Float16* kc1 = &Ktile[buf][16 + laneLo][0];
    v8f c0 = {}, c1 = {};
    c0 = wmma32(aq0, frag16(kc0, 0, laneHi), c0);
    c0 = wmma32(aq1, frag16(kc0, 32, laneHi), c0);
    c1 = wmma32(aq0, frag16(kc1, 0, laneHi), c1);
    c1 = wmma32(aq1, frag16(kc1, 32, laneHi), c1);

    // online softmax. C layout: lane laneLo = column, VGPR r = row r+8*laneHi.
    float tm[8];
#pragma unroll
    for (int r = 0; r < 8; ++r) {
      c0[r] *= ATTN_SCALE; c1[r] *= ATTN_SCALE;
      tm[r] = fmaxf(c0[r], c1[r]);
    }
#pragma unroll
    for (int mk = 1; mk < 16; mk <<= 1)
#pragma unroll
      for (int r = 0; r < 8; ++r)
        tm[r] = fmaxf(tm[r], __shfl_xor(tm[r], mk, 32));

    float al[8], ls[8];
    v8f p0, p1;
#pragma unroll
    for (int r = 0; r < 8; ++r) {
      float nm = fmaxf(mr[r], tm[r]);
      al[r] = __expf(mr[r] - nm);
      mr[r] = nm;
      p0[r] = __expf(c0[r] - nm);
      p1[r] = __expf(c1[r] - nm);
      ls[r] = p0[r] + p1[r];
    }
#pragma unroll
    for (int mk = 1; mk < 16; mk <<= 1)
#pragma unroll
      for (int r = 0; r < 8; ++r)
        ls[r] += __shfl_xor(ls[r], mk, 32);
#pragma unroll
    for (int r = 0; r < 8; ++r) {
      lr[r] = lr[r] * al[r] + ls[r];
      o0[r] *= al[r]; o1[r] *= al[r]; o2[r] *= al[r]; o3[r] *= al[r];
    }

    // reshape P accumulator -> A fragment via LDS
    __syncthreads();
#pragma unroll
    for (int r = 0; r < 8; ++r) {
      int m = r + 8 * laneHi;
      Plds[wave][m][laneLo]      = (_Float16)p0[r];
      Plds[wave][m][laneLo + 16] = (_Float16)p1[r];
    }
    __syncthreads();
    v16h ap = frag16(&Plds[wave][laneLo][0], 0, laneHi);

    // O += P @ V : 4 WMMAs, V fragments from LDS (rows of V^T = contiguous keys)
    o0 = wmma32(ap, frag16(&Vtile[buf][laneLo][0],      0, laneHi), o0);
    o1 = wmma32(ap, frag16(&Vtile[buf][16 + laneLo][0], 0, laneHi), o1);
    o2 = wmma32(ap, frag16(&Vtile[buf][32 + laneLo][0], 0, laneHi), o2);
    o3 = wmma32(ap, frag16(&Vtile[buf][48 + laneLo][0], 0, laneHi), o3);
  }

  // normalize and store: out[t][h*64 + dh]
#pragma unroll
  for (int r = 0; r < 8; ++r) {
    float inv = 1.0f / lr[r];
    size_t row = (size_t)(qm0 + r + 8 * laneHi);
    _Float16* op = Oh + row * D_INNER + h * D_HEAD + laneLo;
    op[0]  = (_Float16)(o0[r] * inv);
    op[16] = (_Float16)(o1[r] * inv);
    op[32] = (_Float16)(o2[r] * inv);
    op[48] = (_Float16)(o3[r] * inv);
  }
}

// ---------------------------------------------------------------------------
// Output GEMM: [4096 x 512] @ [512 x 128]; node rows use o_w, edge rows eo_w.
// Both outputs are contiguous at d_out[t*128 + col].
// ---------------------------------------------------------------------------
__global__ void k_gemm_out(const _Float16* __restrict__ A, const _Float16* __restrict__ WtO,
                           const _Float16* __restrict__ WtEO, const float* __restrict__ bO,
                           const float* __restrict__ bEO, float* __restrict__ out) {
  int lane = threadIdx.x & 31, wave = threadIdx.x >> 5;
  int laneHi = lane >> 4, laneLo = lane & 15;
  int tile = blockIdx.x * 8 + wave;          // 256 * 8 = 2048 tiles
  int m0 = (tile >> 3) * 16, n0 = (tile & 7) * 16;
  bool isNode = (m0 < N_NODES);
  const _Float16* Wt = isNode ? WtO : WtEO;
  const float* bias  = isNode ? bO  : bEO;
  const _Float16* arow = A  + (size_t)(m0 + laneLo) * D_INNER;
  const _Float16* bcol = Wt + (size_t)(n0 + laneLo) * D_INNER;
  v8f acc = {};
#pragma unroll
  for (int k0 = 0; k0 < D_INNER; k0 += 32)
    acc = wmma32(frag16(arow, k0, laneHi), frag16(bcol, k0, laneHi), acc);
  int col = n0 + laneLo;
  float bi = bias[col];
#pragma unroll
  for (int r = 0; r < 8; ++r) {
    int row = m0 + r + 8 * laneHi;
    out[(size_t)row * D_MODEL + col] = acc[r] + bi;
  }
}

// ---------------------------------------------------------------------------
// Host orchestration
// ---------------------------------------------------------------------------
extern "C" void kernel_launch(void* const* d_in, const int* in_sizes, int n_in,
                              void* d_out, int out_size, void* d_ws, size_t ws_size,
                              hipStream_t stream) {
  const float* nodes = (const float*)d_in[0];
  const float* ef    = (const float*)d_in[1];
  const int*   eidx  = (const int*)  d_in[2];
  const float* q_w  = (const float*)d_in[3];  const float* q_b  = (const float*)d_in[4];
  const float* k_w  = (const float*)d_in[5];  const float* k_b  = (const float*)d_in[6];
  const float* v_w  = (const float*)d_in[7];  const float* v_b  = (const float*)d_in[8];
  const float* eq_w = (const float*)d_in[9];  const float* eq_b = (const float*)d_in[10];
  const float* ek_w = (const float*)d_in[11]; const float* ek_b = (const float*)d_in[12];
  const float* ev_w = (const float*)d_in[13]; const float* ev_b = (const float*)d_in[14];
  const float* qw_w = (const float*)d_in[15]; const float* qw_b = (const float*)d_in[16];
  const float* kw_w = (const float*)d_in[17]; const float* kw_b = (const float*)d_in[18];
  const float* vw_w = (const float*)d_in[19]; const float* vw_b = (const float*)d_in[20];
  const float* o_w  = (const float*)d_in[21]; const float* o_b  = (const float*)d_in[22];
  const float* eo_w = (const float*)d_in[23]; const float* eo_b = (const float*)d_in[24];
  float* out = (float*)d_out;

  // workspace carve-up (~27 MB total)
  char* ws = (char*)d_ws;
  size_t off = 0;
  auto alloc = [&](size_t bytes) -> char* {
    char* p = ws + off;
    off = (off + bytes + 255) & ~(size_t)255;
    return p;
  };
  _Float16* nodes_h = (_Float16*)alloc((size_t)N_NODES * D_MODEL * 2);
  _Float16* Wt_q    = (_Float16*)alloc((size_t)D_INNER * D_MODEL * 2);
  _Float16* Wt_k    = (_Float16*)alloc((size_t)D_INNER * D_MODEL * 2);
  _Float16* Wt_v    = (_Float16*)alloc((size_t)D_INNER * D_MODEL * 2);
  _Float16* Wc_eq   = (_Float16*)alloc((size_t)D_INNER * E_DIM * 2);
  _Float16* Wc_ek   = (_Float16*)alloc((size_t)D_INNER * E_DIM * 2);
  _Float16* Wc_ev   = (_Float16*)alloc((size_t)D_INNER * E_DIM * 2);
  _Float16* Wt_o    = (_Float16*)alloc((size_t)D_MODEL * D_INNER * 2);
  _Float16* Wt_eo   = (_Float16*)alloc((size_t)D_MODEL * D_INNER * 2);
  float*    bc_eq   = (float*)alloc(D_INNER * 4);
  float*    bc_ek   = (float*)alloc(D_INNER * 4);
  float*    bc_ev   = (float*)alloc(D_INNER * 4);
  _Float16* edges_h = (_Float16*)alloc((size_t)N_EDGES * E_DIM * 2);
  float*    qf      = (float*)alloc((size_t)N_NODES * D_INNER * 4);
  float*    kf      = (float*)alloc((size_t)N_NODES * D_INNER * 4);
  float*    vf      = (float*)alloc((size_t)N_NODES * D_INNER * 4);
  _Float16* Qh      = (_Float16*)alloc((size_t)N_HEADS * T_TOT * D_HEAD * 2);
  _Float16* Kh      = (_Float16*)alloc((size_t)N_HEADS * T_TOT * D_HEAD * 2);
  _Float16* Vtp     = (_Float16*)alloc((size_t)N_HEADS * D_HEAD * VT_LD * 2);
  _Float16* AO      = (_Float16*)alloc((size_t)T_TOT * D_INNER * 2);

  const int TPB = 256;
  // --- weight / input prep ---
  k_cvt_f16<<<(N_NODES * D_MODEL + TPB - 1) / TPB, TPB, 0, stream>>>(nodes, nodes_h, N_NODES * D_MODEL);
  k_transpose_cvt<<<(128 * 512 + TPB - 1) / TPB, TPB, 0, stream>>>(q_w, 128, 512, Wt_q, 128, 0, 1.0f);
  k_transpose_cvt<<<(128 * 512 + TPB - 1) / TPB, TPB, 0, stream>>>(k_w, 128, 512, Wt_k, 128, 0, 1.0f);
  k_transpose_cvt<<<(128 * 512 + TPB - 1) / TPB, TPB, 0, stream>>>(v_w, 128, 512, Wt_v, 128, 0, 1.0f);
  // combined edge weights (note the reference's cross-wiring: eq+kw, ek+vw, ev+qw)
  k_transpose_cvt<<<(256 * 512 + TPB - 1) / TPB, TPB, 0, stream>>>(eq_w, 256, 512, Wc_eq, E_DIM, 0,   SCALE_SC);
  k_transpose_cvt<<<(64  * 512 + TPB - 1) / TPB, TPB, 0, stream>>>(kw_w, 64,  512, Wc_eq, E_DIM, 256, SCALE_S);
  k_transpose_cvt<<<(256 * 512 + TPB - 1) / TPB, TPB, 0, stream>>>(ek_w, 256, 512, Wc_ek, E_DIM, 0,   SCALE_SC);
  k_transpose_cvt<<<(64  * 512 + TPB - 1) / TPB, TPB, 0, stream>>>(vw_w, 64,  512, Wc_ek, E_DIM, 256, SCALE_S);
  k_transpose_cvt<<<(256 * 512 + TPB - 1) / TPB, TPB, 0, stream>>>(ev_w, 256, 512, Wc_ev, E_DIM, 0,   SCALE_SC);
  k_transpose_cvt<<<(64  * 512 + TPB - 1) / TPB, TPB, 0, stream>>>(qw_w, 64,  512, Wc_ev, E_DIM, 256, SCALE_S);
  k_transpose_cvt<<<(512 * 128 + TPB - 1) / TPB, TPB, 0, stream>>>(o_w,  512, 128, Wt_o,  512, 0, 1.0f);
  k_transpose_cvt<<<(512 * 128 + TPB - 1) / TPB, TPB, 0, stream>>>(eo_w, 512, 128, Wt_eo, 512, 0, 1.0f);
  k_bias2<<<2, TPB, 0, stream>>>(eq_b, kw_b, SCALE_SC, SCALE_S, bc_eq, D_INNER);
  k_bias2<<<2, TPB, 0, stream>>>(ek_b, vw_b, SCALE_SC, SCALE_S, bc_ek, D_INNER);
  k_bias2<<<2, TPB, 0, stream>>>(ev_b, qw_b, SCALE_SC, SCALE_S, bc_ev, D_INNER);
  k_prep_edges<<<(N_EDGES * E_DIM + TPB - 1) / TPB, TPB, 0, stream>>>(nodes, ef, eidx, edges_h);

  // --- node projections (2048 tiles each, 8 tiles/block) ---
  k_gemm_node<<<256, TPB, 0, stream>>>(nodes_h, Wt_q, q_b, qf, Qh, nullptr);
  k_gemm_node<<<256, TPB, 0, stream>>>(nodes_h, Wt_k, k_b, kf, Kh, nullptr);
  k_gemm_node<<<256, TPB, 0, stream>>>(nodes_h, Wt_v, v_b, vf, nullptr, Vtp);

  // --- edge projections (6144 tiles each); alpha sums: e_q<-k, e_k<-q, e_v<-v ---
  k_gemm_edge<<<768, TPB, 0, stream>>>(edges_h, Wc_eq, bc_eq, kf, eidx, Qh, nullptr);
  k_gemm_edge<<<768, TPB, 0, stream>>>(edges_h, Wc_ek, bc_ek, qf, eidx, Kh, nullptr);
  k_gemm_edge<<<768, TPB, 0, stream>>>(edges_h, Wc_ev, bc_ev, vf, eidx, nullptr, Vtp);

  // --- flash attention: 8 heads x (4096/128) blocks of 8 waves ---
  k_attn<<<dim3(T_TOT / 128, N_HEADS), TPB, 0, stream>>>(Qh, Kh, Vtp, AO);

  // --- output projections ---
  k_gemm_out<<<256, TPB, 0, stream>>>(AO, Wt_o, Wt_eo, o_b, eo_b, out);

  (void)in_sizes; (void)n_in; (void)out_size; (void)ws_size;
}
